// MultiHeadAttention_1503238554156
// MI455X (gfx1250) — matrix operand order repaired and checked by reading, not compiled, not run on hardware
//
#include <hip/hip_runtime.h>
#include <hip/hip_bf16.h>

// ---- problem constants (match reference) ----
#define BB 8
#define SS 1024
#define DD 1024
#define HH 16
#define DKK 64

typedef __bf16 bf16_t;
typedef __attribute__((ext_vector_type(16))) __bf16 v16bf;
typedef __attribute__((ext_vector_type(8)))  __bf16 v8bf;
typedef __attribute__((ext_vector_type(8)))  float  v8f;

static __device__ __forceinline__ v8f v8f_zero() {
  v8f z = {0.f, 0.f, 0.f, 0.f, 0.f, 0.f, 0.f, 0.f};
  return z;
}

// =====================================================================
// Elementwise fp32 -> bf16 conversion (one-time pass; halves GEMM bytes)
// =====================================================================
__global__ __launch_bounds__(256) void cvt_f32_bf16(
    const float* __restrict__ src, bf16_t* __restrict__ dst, int n4) {
  const int i = blockIdx.x * blockDim.x + threadIdx.x;
  if (i < n4) {
    const float4 f = ((const float4*)src)[i];
    bf16_t* d = dst + (size_t)i * 4;
    d[0] = (bf16_t)f.x; d[1] = (bf16_t)f.y;
    d[2] = (bf16_t)f.z; d[3] = (bf16_t)f.w;
  }
}

// =====================================================================
// Async-DMA double-buffered bf16 WMMA GEMM:  Y = X @ W^T + bias (scaled)
// X: (8192,1024) bf16 ; W: (1024,1024) bf16 row-major (out_ch major)
// Tiles staged with global_load_async_to_lds_b128 (ASYNCcnt), overlapped
// with v_wmma_f32_16x16x32_bf16 on the other buffer.
// MODE 0: -> Qbf [B,H,S,DK]  (scale=1/sqrt(DK))   MODE 1: -> Kbf [B,H,S,DK]
// MODE 2: -> Vt  [B,H,DK,S]                       MODE 3: -> out [B,S,D] fp32
// Block: 256 threads (8 wave32), tile 64(M) x 128(N), BK=32; each wave
// owns a 32x32 accumulator = 4 WMMAs per K step.
// =====================================================================
template <int MODE>
__global__ __launch_bounds__(256) void proj_gemm(
    const bf16_t* __restrict__ X, const bf16_t* __restrict__ W,
    const float* __restrict__ bias, void* __restrict__ Yout, float scale) {
  __shared__ bf16_t lA[2][64][32];    // 2 x 4 KB
  __shared__ bf16_t lB[2][128][32];   // 2 x 8 KB

  const int m0   = blockIdx.x * 64;
  const int n0   = blockIdx.y * 128;
  const int tid  = threadIdx.x;
  const int lane = tid & 31;
  const int wave = tid >> 5;
  const int wm   = (wave & 1) * 32;   // wave M offset in block tile
  const int wn   = (wave >> 1) * 32;  // wave N offset in block tile
  const int nl   = lane & 15;
  const int hi   = lane >> 4;

  // per-thread async staging assignment (16B A + 32B B per K step)
  const int arow = tid >> 2;          // 0..63
  const int aseg = (tid & 3) * 8;     // bf16 elems: 0,8,16,24
  const int brow = tid >> 1;          // 0..127
  const int bseg = (tid & 1) * 16;    // bf16 elems: 0,16

  const bf16_t* gA = X + (size_t)(m0 + arow) * DD + aseg;
  const bf16_t* gB = W + (size_t)(n0 + brow) * DD + bseg;

  auto issue = [&](int buf, int k0) {
    const unsigned la = (unsigned)(size_t)&lA[buf][arow][aseg];
    const unsigned lb = (unsigned)(size_t)&lB[buf][brow][bseg];
    const unsigned long long ga = (unsigned long long)(size_t)(gA + k0);
    const unsigned long long gb = (unsigned long long)(size_t)(gB + k0);
    asm volatile("global_load_async_to_lds_b128 %0, %1, off"
                 :: "v"(la), "v"(ga) : "memory");
    asm volatile("global_load_async_to_lds_b128 %0, %1, off"
                 :: "v"(lb), "v"(gb) : "memory");
    asm volatile("global_load_async_to_lds_b128 %0, %1, off"
                 :: "v"(lb + 16u), "v"(gb + 16ull) : "memory");
  };

  v8f acc[2][2];
#pragma unroll
  for (int i = 0; i < 2; i++)
#pragma unroll
    for (int j = 0; j < 2; j++) acc[i][j] = v8f_zero();

  issue(0, 0);  // prime buffer 0

  for (int it = 0; it < DD / 32; ++it) {
    const int buf = it & 1;
    // wait for this buffer's DMAs (per wave), then make it WG-visible
    asm volatile("s_wait_asynccnt 0x0" ::: "memory");
    __syncthreads();
    // overlap: kick off next buffer's DMAs before computing
    if (it + 1 < DD / 32) issue(buf ^ 1, (it + 1) * 32);

    // ---- 4 WMMAs per wave; A and B fragments both use contiguous k = hi*16+s ----
#pragma unroll
    for (int im = 0; im < 2; im++) {
      union { v16bf v; v8bf h[2]; } afr;
      const int ar = wm + im * 16 + nl;                     // A row = lane&15
      afr.h[0] = *(const v8bf*)&lA[buf][ar][hi * 16];       // K = hi*16+0..7
      afr.h[1] = *(const v8bf*)&lA[buf][ar][hi * 16 + 8];   // K = hi*16+8..15
#pragma unroll
      for (int in = 0; in < 2; in++) {
        const int bn = wn + in * 16 + nl;                   // B col = lane&15
        v16bf bfr = *(const v16bf*)&lB[buf][bn][hi * 16];   // K = hi*16+0..15
        acc[im][in] = __builtin_amdgcn_wmma_f32_16x16x32_bf16(
            false, afr.v, false, bfr, (short)0, acc[im][in], false, false);
      }
    }
    // reads of lA/lB[buf] are dscnt-complete before this wave passes the
    // next iteration's s_wait_asynccnt + barrier, so no trailing barrier.
  }

  // ---- epilogue: bias + scale + layout-specific store ----
#pragma unroll
  for (int im = 0; im < 2; im++) {
#pragma unroll
    for (int in = 0; in < 2; in++) {
      const int col = n0 + wn + in * 16 + nl;     // 0..1023
      const float bv = bias[col];
#pragma unroll
      for (int r = 0; r < 8; r++) {
        const int row = m0 + wm + im * 16 + hi * 8 + r;  // 0..8191
        const float y = (acc[im][in][r] + bv) * scale;
        if (MODE == 3) {
          ((float*)Yout)[(size_t)row * DD + col] = y;
        } else {
          const int b_ = row >> 10;
          const int s_ = row & (SS - 1);
          const int h_ = col >> 6;
          const int dk = col & (DKK - 1);
          bf16_t* Y = (bf16_t*)Yout;
          if (MODE == 2) {  // Vt: [B,H,DK,S]
            Y[(((size_t)b_ * HH + h_) * DKK + dk) * SS + s_] = (bf16_t)y;
          } else {          // Q/K: [B,H,S,DK]
            Y[(((size_t)b_ * HH + h_) * SS + s_) * DKK + dk] = (bf16_t)y;
          }
        }
      }
    }
  }
}

// =====================================================================
// Flash-style fused attention: softmax(mask(Q K^T / sqrt(dk))) @ V
// Grid: (B*H, S/128); block 256 = 8 waves; each wave owns 16 query rows
// and streams all 1024 keys in 32-key steps (online softmax).
// Per step: 4 WMMA (scores) + 4 WMMA (P@V). Scale folded into Q.
// =====================================================================
__global__ __launch_bounds__(256) void attn_kernel(
    const bf16_t* __restrict__ Qbf, const bf16_t* __restrict__ Kbf,
    const bf16_t* __restrict__ Vt, const int* __restrict__ mask,
    bf16_t* __restrict__ O) {
  __shared__ bf16_t Pl[8][16][32];  // per-wave P tile, 8 KB total

  const int bh   = blockIdx.x;       // 0..127
  const int b_   = bh >> 4;
  const int h_   = bh & (HH - 1);
  const int wave = threadIdx.x >> 5;
  const int lane = threadIdx.x & 31;
  const int nl   = lane & 15;
  const int hi   = lane >> 4;
  const int q0   = blockIdx.y * 128 + wave * 16;

  const bf16_t* Qh = Qbf + (((size_t)b_ * HH + h_) * SS) * DKK;
  const bf16_t* Kh = Kbf + (((size_t)b_ * HH + h_) * SS) * DKK;
  const bf16_t* Vh = Vt  + (((size_t)b_ * HH + h_) * DKK) * SS;
  const int* Mb    = mask + (size_t)b_ * SS * SS;

  // Q A-fragments for both 32-wide K chunks of DK=64 (held in registers)
  union { v16bf v; v8bf h[2]; } qa[2];
#pragma unroll
  for (int c = 0; c < 2; c++) {
    const bf16_t* qr = Qh + (size_t)(q0 + nl) * DKK + c * 32;
    qa[c].h[0] = *(const v8bf*)(qr + hi * 16);
    qa[c].h[1] = *(const v8bf*)(qr + hi * 16 + 8);
  }

  v8f out[4];
#pragma unroll
  for (int t = 0; t < 4; t++) out[t] = v8f_zero();
  float mrun[8], lrun[8];
#pragma unroll
  for (int r = 0; r < 8; r++) { mrun[r] = -3.0e38f; lrun[r] = 0.f; }

  for (int j = 0; j < SS; j += 32) {
    // ---- scores: two 16x16 tiles over keys [j,j+16) and [j+16,j+32) ----
    v8f s0 = v8f_zero(), s1 = v8f_zero();
#pragma unroll
    for (int c = 0; c < 2; c++) {
      v16bf kb0 = *(const v16bf*)(Kh + (size_t)(j + nl) * DKK + c * 32 + hi * 16);
      v16bf kb1 = *(const v16bf*)(Kh + (size_t)(j + 16 + nl) * DKK + c * 32 + hi * 16);
      s0 = __builtin_amdgcn_wmma_f32_16x16x32_bf16(false, qa[c].v, false, kb0,
                                                   (short)0, s0, false, false);
      s1 = __builtin_amdgcn_wmma_f32_16x16x32_bf16(false, qa[c].v, false, kb1,
                                                   (short)0, s1, false, false);
    }

    // ---- mask + online softmax (row = hi*8 + r spans 16 lanes of a half) ----
#pragma unroll
    for (int r = 0; r < 8; r++) {
      const int row = q0 + hi * 8 + r;
      const int* mr = Mb + (size_t)row * SS + j;
      float v0 = (mr[nl] == 0)      ? -1e9f : s0[r];
      float v1 = (mr[16 + nl] == 0) ? -1e9f : s1[r];
      float rmax = fmaxf(v0, v1);
#pragma unroll
      for (int off = 1; off < 16; off <<= 1)
        rmax = fmaxf(rmax, __shfl_xor(rmax, off, 32));
      const float mnew  = fmaxf(mrun[r], rmax);
      const float alpha = __expf(mrun[r] - mnew);
      const float p0 = __expf(v0 - mnew);
      const float p1 = __expf(v1 - mnew);
      float rs = p0 + p1;
#pragma unroll
      for (int off = 1; off < 16; off <<= 1) rs += __shfl_xor(rs, off, 32);
      lrun[r] = lrun[r] * alpha + rs;
      mrun[r] = mnew;
#pragma unroll
      for (int t = 0; t < 4; t++) out[t][r] *= alpha;
      Pl[wave][hi * 8 + r][nl]      = (bf16_t)p0;
      Pl[wave][hi * 8 + r][16 + nl] = (bf16_t)p1;
    }
    __builtin_amdgcn_wave_barrier();
    asm volatile("s_wait_dscnt 0x0" ::: "memory");  // wave-local C->A relayout

    // ---- P (16x32) A-fragment, then out += P @ V over 4 dk column tiles ----
    union { v16bf v; v8bf h[2]; } pa;
    pa.h[0] = *(const v8bf*)&Pl[wave][nl][hi * 16];
    pa.h[1] = *(const v8bf*)&Pl[wave][nl][hi * 16 + 8];
#pragma unroll
    for (int t = 0; t < 4; t++) {
      v16bf vb = *(const v16bf*)(Vh + (size_t)(t * 16 + nl) * SS + j + hi * 16);
      out[t] = __builtin_amdgcn_wmma_f32_16x16x32_bf16(false, pa.v, false, vb,
                                                       (short)0, out[t], false, false);
    }
    __builtin_amdgcn_wave_barrier();
  }

  // ---- normalize and store O [B,S,D] bf16 ----
#pragma unroll
  for (int t = 0; t < 4; t++) {
#pragma unroll
    for (int r = 0; r < 8; r++) {
      const int row = q0 + hi * 8 + r;
      const float y = out[t][r] / lrun[r];
      O[((size_t)b_ * SS + row) * DD + h_ * DKK + t * 16 + nl] = (bf16_t)y;
    }
  }
}

// =====================================================================
extern "C" void kernel_launch(void* const* d_in, const int* in_sizes, int n_in,
                              void* d_out, int out_size, void* d_ws,
                              size_t ws_size, hipStream_t stream) {
  (void)in_sizes; (void)n_in; (void)out_size; (void)ws_size;
  const float* q    = (const float*)d_in[0];
  const float* k    = (const float*)d_in[1];
  const float* v    = (const float*)d_in[2];
  const int*   mask = (const int*)d_in[3];
  const float* WQ_w = (const float*)d_in[4];
  const float* WQ_b = (const float*)d_in[5];
  const float* WK_w = (const float*)d_in[6];
  const float* WK_b = (const float*)d_in[7];
  const float* WV_w = (const float*)d_in[8];
  const float* WV_b = (const float*)d_in[9];
  const float* WO_w = (const float*)d_in[10];
  const float* WO_b = (const float*)d_in[11];

  char* ws = (char*)d_ws;
  const size_t actSz = (size_t)BB * SS * DD * sizeof(bf16_t);  // 16 MiB
  const size_t wSz   = (size_t)DD * DD * sizeof(bf16_t);       //  2 MiB
  bf16_t* Qbf = (bf16_t*)(ws + 0 * actSz);
  bf16_t* Kbf = (bf16_t*)(ws + 1 * actSz);
  bf16_t* Vt  = (bf16_t*)(ws + 2 * actSz);
  bf16_t* Xq  = (bf16_t*)(ws + 3 * actSz);
  bf16_t* O   = Xq;  // alias: Xq is dead after the Q projection
  bf16_t* Xk  = (bf16_t*)(ws + 4 * actSz);
  bf16_t* Xv  = (bf16_t*)(ws + 5 * actSz);
  bf16_t* Wqb = (bf16_t*)(ws + 6 * actSz + 0 * wSz);
  bf16_t* Wkb = (bf16_t*)(ws + 6 * actSz + 1 * wSz);
  bf16_t* Wvb = (bf16_t*)(ws + 6 * actSz + 2 * wSz);
  bf16_t* Wob = (bf16_t*)(ws + 6 * actSz + 3 * wSz);

  // one-time fp32 -> bf16 conversions (vectorized x4)
  const int nAct4 = (BB * SS * DD) / 4;   // 2,097,152
  const int nW4   = (DD * DD) / 4;        //   262,144
  cvt_f32_bf16<<<nAct4 / 256, 256, 0, stream>>>(q, Xq, nAct4);
  cvt_f32_bf16<<<nAct4 / 256, 256, 0, stream>>>(k, Xk, nAct4);
  cvt_f32_bf16<<<nAct4 / 256, 256, 0, stream>>>(v, Xv, nAct4);
  cvt_f32_bf16<<<nW4 / 256, 256, 0, stream>>>(WQ_w, Wqb, nW4);
  cvt_f32_bf16<<<nW4 / 256, 256, 0, stream>>>(WK_w, Wkb, nW4);
  cvt_f32_bf16<<<nW4 / 256, 256, 0, stream>>>(WV_w, Wvb, nW4);
  cvt_f32_bf16<<<nW4 / 256, 256, 0, stream>>>(WO_w, Wob, nW4);

  const dim3 g(128, 8), blk(256);
  // scale 1/sqrt(DK)=0.125 folded into Q projection epilogue
  proj_gemm<0><<<g, blk, 0, stream>>>(Xq, Wqb, WQ_b, Qbf, 0.125f);
  proj_gemm<1><<<g, blk, 0, stream>>>(Xk, Wkb, WK_b, Kbf, 1.0f);
  proj_gemm<2><<<g, blk, 0, stream>>>(Xv, Wvb, WV_b, Vt, 1.0f);
  attn_kernel<<<dim3(BB * HH, SS / 128), blk, 0, stream>>>(Qbf, Kbf, Vt, mask, O);
  proj_gemm<3><<<g, blk, 0, stream>>>(O, Wob, WO_b, d_out, 1.0f);
}
